// MultiheadAttentionUnifiedLiT_23381801959858
// MI455X (gfx1250) — matrix operand order repaired and checked by reading, not compiled, not run on hardware
//
#include <hip/hip_runtime.h>
#include <hip/hip_bf16.h>
#include <math.h>

// ---------------------------------------------------------------------------
// MI455X (gfx1250) multi-head attention, bf16 WMMA pipeline.
// B=2 S=2048 D=1024 H=16 Hd=64.  d_out = [out f32 (B*S*D)] ++ [attn f32 (B*H*S*S)]
// ---------------------------------------------------------------------------

typedef __attribute__((ext_vector_type(16))) __bf16 v16bf;
typedef __attribute__((ext_vector_type(8)))  float  v8f;
typedef unsigned short ushort_t;

union FragB16 {
    v16bf    v;
    unsigned u[8];
    ushort_t s[16];
};

// round-to-nearest-even f32 -> bf16 (bit pattern as ushort)
__device__ __forceinline__ ushort_t f2bf(float f) {
    union { float f; unsigned u; } x; x.f = f;
    unsigned r = x.u + 0x7FFFu + ((x.u >> 16) & 1u);
    return (ushort_t)(r >> 16);
}

__device__ __forceinline__ unsigned pack2bf(float lo, float hi) {
    return (unsigned)f2bf(lo) | ((unsigned)f2bf(hi) << 16);
}

__device__ __forceinline__ v8f wmma_bf16(const FragB16& a, const FragB16& b, v8f c) {
    // v_wmma_f32_16x16x32_bf16  D = A(16x32) * B(32x16) + C
    return __builtin_amdgcn_wmma_f32_16x16x32_bf16(
        /*neg_a=*/false, a.v, /*neg_b=*/false, b.v,
        /*c_mod=*/(short)0, c, /*reuse_a=*/false, /*reuse_b=*/false);
}

// ---------------------------------------------------------------------------
// Generic GEMM:  out[m,n] = sum_k X[m,k] * W[n,k] + bias[n]
//   XF32:  X is f32 (converted to bf16 on load); else X is bf16
//   OUTF32: store f32 row-major (+bias) ; else store bf16 head-split [B,H,S,64]
// 64x64 workgroup tile, 256 threads = 8 waves, 2 WMMA accumulators per wave.
// ---------------------------------------------------------------------------
template <bool XF32, bool OUTF32>
__global__ __launch_bounds__(256)
void mha_gemm_kernel(const float* __restrict__ Xf, const ushort_t* __restrict__ Xb,
                     const float* __restrict__ W,  const float* __restrict__ bias,
                     ushort_t* __restrict__ out_bf, float* __restrict__ out_f,
                     int M, int N, int K)
{
    __shared__ ushort_t Xs[64][32];
    __shared__ ushort_t Ws[64][32];

    const int tid  = threadIdx.x;
    const int lane = tid & 31;
    const int wave = tid >> 5;
    const int m0 = blockIdx.x * 64;
    const int n0 = blockIdx.y * 64;

    const int m_sub  = wave & 3;          // which 16-row subtile
    const int n_sub0 = (wave >> 2) * 2;   // first of two 16-col subtiles

    v8f acc0 = {}; v8f acc1 = {};

    const int am   = m_sub * 16 + (lane & 15);
    const int koff = (lane >> 4) * 8;
    const int kb   = (lane >> 4) * 16;
    const int bn0  = n_sub0 * 16 + (lane & 15);
    const int bn1  = bn0 + 16;

    for (int k0 = 0; k0 < K; k0 += 32) {
        __syncthreads();
        // cooperative tile load: 2048 elems per tile, float4/uint2 per thread x2
        #pragma unroll
        for (int it = 0; it < 2; ++it) {
            int e = (tid + it * 256) * 4;     // element index in 64x32 tile
            int r = e >> 5, c = e & 31;
            uint2 p;
            if (XF32) {
                float4 x = *(const float4*)&Xf[(size_t)(m0 + r) * K + k0 + c];
                p.x = pack2bf(x.x, x.y);
                p.y = pack2bf(x.z, x.w);
            } else {
                p = *(const uint2*)&Xb[(size_t)(m0 + r) * K + k0 + c];
            }
            *(uint2*)&Xs[r][c] = p;
            float4 w = *(const float4*)&W[(size_t)(n0 + r) * K + k0 + c];
            uint2 q;
            q.x = pack2bf(w.x, w.y);
            q.y = pack2bf(w.z, w.w);
            *(uint2*)&Ws[r][c] = q;
        }
        __syncthreads();

        FragB16 a, b0, b1;
        #pragma unroll
        for (int j = 0; j < 8; ++j) {
            int kk = kb + 2 * j;
            a.u[j] = *(const unsigned*)&Xs[am][kk];
        }
        #pragma unroll
        for (int j = 0; j < 8; ++j) {
            b0.u[j] = *(const unsigned*)&Ws[bn0][kb + 2 * j];
            b1.u[j] = *(const unsigned*)&Ws[bn1][kb + 2 * j];
        }
        acc0 = wmma_bf16(a, b0, acc0);
        acc1 = wmma_bf16(a, b1, acc1);
    }

    // C/D layout: VGPR r -> M = r + 8*(lane>=16); N = lane&15 within subtile
    const int mbase = m0 + m_sub * 16 + (lane >> 4) * 8;
    const int ncol0 = n0 + n_sub0 * 16 + (lane & 15);
    const int ncol1 = ncol0 + 16;
    const float bi0 = bias[ncol0];
    const float bi1 = bias[ncol1];

    #pragma unroll
    for (int r = 0; r < 8; ++r) {
        int row = mbase + r;
        float v0 = acc0[r] + bi0;
        float v1 = acc1[r] + bi1;
        if (OUTF32) {
            out_f[(size_t)row * N + ncol0] = v0;
            out_f[(size_t)row * N + ncol1] = v1;
        } else {
            // head-split bf16: row=b*2048+s, col=h*64+hd -> [(b*16+h)*2048+s]*64+hd
            int b_ = row >> 11, s_ = row & 2047;
            int h0 = ncol0 >> 6, hd0 = ncol0 & 63;
            int h1 = ncol1 >> 6, hd1 = ncol1 & 63;
            out_bf[((size_t)(b_ * 16 + h0) * 2048 + s_) * 64 + hd0] = f2bf(v0);
            out_bf[((size_t)(b_ * 16 + h1) * 2048 + s_) * 64 + hd1] = f2bf(v1);
        }
    }
}

// ---------------------------------------------------------------------------
// Scores + softmax: one WG per (b,h,16 query rows).
// Each of 8 waves owns a 256-key chunk: 16 n-subtiles x (Hd=64 -> 2 k-steps).
// Full 16x2048 logit block lives in 128KB dynamic LDS (WGP has 320KB).
// Probabilities written f32 (b128 stores) directly to d_out attention region.
// ---------------------------------------------------------------------------
__global__ __launch_bounds__(256)
void mha_scores_kernel(const ushort_t* __restrict__ Qb, const ushort_t* __restrict__ Kb,
                       float* __restrict__ attn)
{
    extern __shared__ float logits[];   // [16][2048]
    const int tid  = threadIdx.x;
    const int lane = tid & 31;
    const int wave = tid >> 5;
    const int q0 = blockIdx.x * 16;
    const int bh = blockIdx.z * 16 + blockIdx.y;
    const size_t base = (size_t)bh * 2048 * 64;

    // A fragments: Q rows q0..q0+15, two k-steps over Hd=64 (shared by all tiles)
    FragB16 a[2];
    const int am   = lane & 15;
    const int koff = (lane >> 4) * 8;
    #pragma unroll
    for (int ks = 0; ks < 2; ++ks) {
        const ushort_t* qrow = Qb + base + (size_t)(q0 + am) * 64 + ks * 32;
        #pragma unroll
        for (int j = 0; j < 8; ++j) {
            int kk = 2 * koff + 2 * j;
            a[ks].u[j] = *(const unsigned*)&qrow[kk];
        }
    }

    const int kb = (lane >> 4) * 16;
    for (int t = 0; t < 16; ++t) {
        int key0 = wave * 256 + t * 16;
        v8f acc = {};
        #pragma unroll
        for (int ks = 0; ks < 2; ++ks) {
            FragB16 b;
            // B(k=d, n=key): lane owns one key row; 16 contiguous bf16 of d
            const ushort_t* krow = Kb + base + (size_t)(key0 + (lane & 15)) * 64 + ks * 32 + kb;
            #pragma unroll
            for (int j = 0; j < 8; ++j) b.u[j] = *(const unsigned*)&krow[2 * j];
            acc = wmma_bf16(a[ks], b, acc);
        }
        const int mb = (lane >> 4) * 8;
        const int nn = key0 + (lane & 15);
        #pragma unroll
        for (int r = 0; r < 8; ++r)
            logits[(size_t)(mb + r) * 2048 + nn] = acc[r] * 0.125f;  // 1/sqrt(64)
    }
    __syncthreads();

    // softmax: each wave handles 2 rows, float4-vectorized (wave32 shuffles)
    #pragma unroll
    for (int rr = 0; rr < 2; ++rr) {
        int row = wave * 2 + rr;
        float4* lrow4 = (float4*)(logits + (size_t)row * 2048);   // 512 float4
        float mx = -1e30f;
        for (int i = lane; i < 512; i += 32) {
            float4 v = lrow4[i];
            mx = fmaxf(mx, fmaxf(fmaxf(v.x, v.y), fmaxf(v.z, v.w)));
        }
        #pragma unroll
        for (int off = 16; off; off >>= 1) mx = fmaxf(mx, __shfl_xor(mx, off, 32));
        float sum = 0.f;
        for (int i = lane; i < 512; i += 32) {
            float4 v = lrow4[i];
            v.x = __expf(v.x - mx); v.y = __expf(v.y - mx);
            v.z = __expf(v.z - mx); v.w = __expf(v.w - mx);
            lrow4[i] = v;
            sum += (v.x + v.y) + (v.z + v.w);
        }
        #pragma unroll
        for (int off = 16; off; off >>= 1) sum += __shfl_xor(sum, off, 32);
        float inv = 1.0f / sum;
        float4* orow4 = (float4*)(attn + ((size_t)bh * 2048 + (q0 + row)) * 2048);
        for (int i = lane; i < 512; i += 32) {
            float4 v = lrow4[i];
            v.x *= inv; v.y *= inv; v.z *= inv; v.w *= inv;
            orow4[i] = v;
        }
    }
}

// ---------------------------------------------------------------------------
// AV: Y[b,h,q,:] = P[b,h,q,:] @ V[b,h,:,:].  64 q-rows x 64 d-cols per WG.
// Probs read f32 (b128) from d_out -> bf16 LDS; V tile copied global->LDS with
// CDNA5 async-LDS loads (ASYNCcnt).  Y stored bf16 row-major [B*S, D].
// ---------------------------------------------------------------------------
__global__ __launch_bounds__(256)
void mha_av_kernel(const float* __restrict__ attn, const ushort_t* __restrict__ Vb,
                   ushort_t* __restrict__ Yb)
{
    __shared__ ushort_t Ps[64][32];
    __shared__ ushort_t Vs[32][64];

    const int tid  = threadIdx.x;
    const int lane = tid & 31;
    const int wave = tid >> 5;
    const int q0 = blockIdx.x * 64;
    const int h  = blockIdx.y;
    const int b  = blockIdx.z;
    const int bh = b * 16 + h;
    const size_t abase = (size_t)bh * 2048 * 2048;
    const size_t vbase = (size_t)bh * 2048 * 64;

    const int m_sub  = wave & 3;
    const int n_sub0 = (wave >> 2) * 2;

    v8f acc0 = {}, acc1 = {};

    const int am   = m_sub * 16 + (lane & 15);
    const int koff = (lane >> 4) * 8;
    const int kb   = (lane >> 4) * 16;
    const int bn0  = n_sub0 * 16 + (lane & 15);
    const int bn1  = bn0 + 16;

    // per-thread addresses for the async V-tile copy (16B per thread = 4KB tile)
    const int vr = tid >> 3;            // 0..31  (key row within tile)
    const int vc = (tid & 7) * 8;       // 0..56  (d column, 8 bf16 = 16B)
    const unsigned vs_lds = (unsigned)(unsigned long long)&Vs[vr][vc]; // LDS addr = low 32 bits

    for (int k0 = 0; k0 < 2048; k0 += 32) {
        // kick off async global->LDS copy of the V tile (ASYNCcnt-tracked)
        {
            const ushort_t* gsrc = Vb + vbase + (size_t)(k0 + vr) * 64 + vc;
            asm volatile("global_load_async_to_lds_b128 %0, %1, off"
                         :: "v"(vs_lds), "v"(gsrc) : "memory");
        }
        // prefetch next k-step's probability rows while we convert this one
        if (k0 + 32 < 2048)
            __builtin_prefetch(&attn[abase + (size_t)(q0 + (tid >> 5)) * 2048 + k0 + 32], 0, 0);

        __syncthreads();
        // probs: f32 -> bf16, float4 per thread x2
        #pragma unroll
        for (int it = 0; it < 2; ++it) {
            int e = (tid + it * 256) * 4;
            int r = e >> 5, c = e & 31;
            float4 x = *(const float4*)&attn[abase + (size_t)(q0 + r) * 2048 + k0 + c];
            uint2 p;
            p.x = pack2bf(x.x, x.y);
            p.y = pack2bf(x.z, x.w);
            *(uint2*)&Ps[r][c] = p;
        }
        asm volatile("s_wait_asynccnt 0x0" ::: "memory");
        __syncthreads();

        FragB16 a, b0, b1;
        #pragma unroll
        for (int j = 0; j < 8; ++j) {
            int kk = kb + 2 * j;
            a.u[j] = *(const unsigned*)&Ps[am][kk];
        }
        #pragma unroll
        for (int j = 0; j < 8; ++j) {
            unsigned lo0 = Vs[kb + 2 * j][bn0], hi0 = Vs[kb + 2 * j + 1][bn0];
            unsigned lo1 = Vs[kb + 2 * j][bn1], hi1 = Vs[kb + 2 * j + 1][bn1];
            b0.u[j] = lo0 | (hi0 << 16);
            b1.u[j] = lo1 | (hi1 << 16);
        }
        acc0 = wmma_bf16(a, b0, acc0);
        acc1 = wmma_bf16(a, b1, acc1);
    }

    const int mbase = q0 + m_sub * 16 + (lane >> 4) * 8;
    const int d0 = n_sub0 * 16 + (lane & 15);
    const int d1 = d0 + 16;
    #pragma unroll
    for (int r = 0; r < 8; ++r) {
        int row = b * 2048 + mbase + r;
        Yb[(size_t)row * 1024 + h * 64 + d0] = f2bf(acc0[r]);
        Yb[(size_t)row * 1024 + h * 64 + d1] = f2bf(acc1[r]);
    }
}

// ---------------------------------------------------------------------------
extern "C" void kernel_launch(void* const* d_in, const int* in_sizes, int n_in,
                              void* d_out, int out_size, void* d_ws, size_t ws_size,
                              hipStream_t stream) {
    const float* query = (const float*)d_in[0];
    const float* key_  = (const float*)d_in[1];
    const float* value = (const float*)d_in[2];
    const float* q_w = (const float*)d_in[3];  const float* q_b = (const float*)d_in[4];
    const float* k_w = (const float*)d_in[5];  const float* k_b = (const float*)d_in[6];
    const float* v_w = (const float*)d_in[7];  const float* v_b = (const float*)d_in[8];
    const float* o_w = (const float*)d_in[9];  const float* o_b = (const float*)d_in[10];

    const size_t BSD = (size_t)2 * 2048 * 1024;      // 4,194,304
    float* out  = (float*)d_out;                     // [B,S,D]
    float* attn = out + BSD;                         // [B,H,S,S]

    ushort_t* Qb = (ushort_t*)d_ws;                  // bf16 [B,H,S,64]
    ushort_t* Kb = Qb + BSD;
    ushort_t* Vb = Kb + BSD;
    ushort_t* Yb = Vb + BSD;                         // bf16 [B*S, D]

    dim3 blk(256);
    dim3 gGemm(4096 / 64, 1024 / 64);                // 64 x 16 tiles

    // in-projections (f32 input -> bf16 head-split)
    mha_gemm_kernel<true,  false><<<gGemm, blk, 0, stream>>>(query, nullptr, q_w, q_b, Qb, nullptr, 4096, 1024, 1024);
    mha_gemm_kernel<true,  false><<<gGemm, blk, 0, stream>>>(key_,  nullptr, k_w, k_b, Kb, nullptr, 4096, 1024, 1024);
    mha_gemm_kernel<true,  false><<<gGemm, blk, 0, stream>>>(value, nullptr, v_w, v_b, Vb, nullptr, 4096, 1024, 1024);

    // QK^T/sqrt(d) + softmax -> attention probs (f32, straight to d_out)
    mha_scores_kernel<<<dim3(128, 16, 2), blk, 16 * 2048 * sizeof(float), stream>>>(Qb, Kb, attn);

    // P @ V -> Y (bf16)
    mha_av_kernel<<<dim3(32, 16, 2), blk, 0, stream>>>(attn, Vb, Yb);

    // out-projection (bf16 input -> f32 output + bias)
    mha_gemm_kernel<false, true><<<gGemm, blk, 0, stream>>>(nullptr, Yb, o_w, o_b, nullptr, out, 4096, 1024, 1024);
}
